// MixtureOfBlockAttention_63350767616595
// MI455X (gfx1250) — matrix operand order repaired and checked by reading, not compiled, not run on hardware
//
#include <hip/hip_runtime.h>
#include <cstdint>

typedef __attribute__((ext_vector_type(16))) __bf16 bf16x16;
typedef __attribute__((ext_vector_type(8)))  __bf16 bf16x8;
typedef __attribute__((ext_vector_type(8)))  float  f32x8;
typedef __attribute__((ext_vector_type(4)))  float  f32x4;
typedef int vint4 __attribute__((vector_size(16)));

#define S_LEN 4096
#define DIM_  1024
#define NHEAD 16
#define HDIM  64
#define BLKSZ 512
#define NBLK  8
#define TOPK_ 3
#define MROWS 8192  // B*S

// ---------------- optional CDNA5 async global->LDS path ----------------------

#if defined(__has_builtin)
#if __has_builtin(__builtin_amdgcn_global_load_async_to_lds_b128) && \
    __has_builtin(__builtin_amdgcn_s_wait_asynccnt)
#define HAVE_ASYNC 1
#endif
#endif
#ifndef HAVE_ASYNC
#define HAVE_ASYNC 0
#endif

__device__ __forceinline__ void g2l_b128(const __bf16* g, __bf16* l) {
#if HAVE_ASYNC
  __builtin_amdgcn_global_load_async_to_lds_b128(
      (__attribute__((address_space(1))) vint4*)(uintptr_t)g,
      (__attribute__((address_space(3))) vint4*)l, 0, 0);
#else
  *(uint4*)l = *(const uint4*)g;
#endif
}

__device__ __forceinline__ void async_join() {
#if HAVE_ASYNC
  __builtin_amdgcn_s_wait_asynccnt(0);
#endif
}

// ---------------- WMMA helpers (layouts per CDNA5 ISA 7.12.2) ----------------

__device__ __forceinline__ f32x8 wmma_bf16(bf16x16 a, bf16x16 b, f32x8 c) {
  return __builtin_amdgcn_wmma_f32_16x16x32_bf16(false, a, false, b, (short)0, c,
                                                 false, false);
}

// A-matrix 16x32 frag: element i <-> K=(i&7)+8*(lane>=16)+16*(i>=8); M=lane&15.
__device__ __forceinline__ bf16x16 load_frag_a(const __bf16* base, int ld, int lane) {
  const __bf16* row = base + (lane & 15) * ld + ((lane & 16) ? 8 : 0);
  bf16x16 f;
#pragma unroll
  for (int i = 0; i < 8; ++i) { f[i] = row[i]; f[i + 8] = row[16 + i]; }
  return f;
}

// B-matrix 32x16 frag from N-major ([N][K]) tile: element i <-> K=(i&7)+8*(lane>=16)+16*(i>=8).
__device__ __forceinline__ bf16x16 load_frag_b(const __bf16* baseT, int ld, int lane) {
  const __bf16* row = baseT + (lane & 15) * ld + ((lane & 16) ? 8 : 0);
  bf16x16 f;
#pragma unroll
  for (int i = 0; i < 8; ++i) { f[i] = row[i]; f[i + 8] = row[16 + i]; }
  return f;
}

// ---------------- elementwise convert (vectorized b128 in/out) ---------------

__global__ void f32_to_bf16_kernel(const float* __restrict__ in,
                                   __bf16* __restrict__ out, int n) {
  int i = (blockIdx.x * blockDim.x + threadIdx.x) * 8;
  if (i >= n) return;
  f32x4 a = *(const f32x4*)&in[i];
  f32x4 b = *(const f32x4*)&in[i + 4];
  bf16x8 o;
#pragma unroll
  for (int e = 0; e < 4; ++e) { o[e] = (__bf16)a[e]; o[e + 4] = (__bf16)b[e]; }
  *(bf16x8*)&out[i] = o;
}

// ------ tiled WMMA GEMM, double-buffered LDS: C f32 = A bf16 * B bf16 --------

__global__ __launch_bounds__(256) void gemm_bf16_kernel(
    const __bf16* __restrict__ A, const __bf16* __restrict__ Bm,
    float* __restrict__ C, int M, int N, int K) {
  __shared__ __align__(16) __bf16 As[2][128][40];  // [M][K], pad 8
  __shared__ __align__(16) __bf16 Bs[2][128][40];  // transposed [N][K]
  const int tid = threadIdx.x;
  const int lane = tid & 31, wave = tid >> 5;
  const int m0 = blockIdx.x * 128, n0 = blockIdx.y * 128;
  const int wm = (wave & 3) * 32;   // 2 M-subtiles of 16
  const int wn = (wave >> 2) * 64;  // 4 N-subtiles of 16
  const int nk = K >> 5;
  f32x8 acc[2][4] = {};

  int arow[2], aoff[2], bkk[2], bnn[2];
#pragma unroll
  for (int u = 0; u < 2; ++u) {
    int c = tid + u * 256;
    arow[u] = c >> 2;  aoff[u] = (c & 3) * 8;   // A: 128 rows x 4 chunks
    bkk[u] = c >> 4;   bnn[u] = (c & 15) * 8;   // B: 32 k-rows x 16 chunks
  }

  bf16x8 breg[2];
  auto issueA = [&](int buf, int k0) {
#pragma unroll
    for (int u = 0; u < 2; ++u)
      g2l_b128(&A[(size_t)(m0 + arow[u]) * K + k0 + aoff[u]],
               &As[buf][arow[u]][aoff[u]]);
  };
  auto loadB = [&](int k0) {
#pragma unroll
    for (int u = 0; u < 2; ++u)
      breg[u] = *(const bf16x8*)&Bm[(size_t)(k0 + bkk[u]) * N + n0 + bnn[u]];
  };
  auto scatterB = [&](int buf) {
#pragma unroll
    for (int u = 0; u < 2; ++u)
#pragma unroll
      for (int e = 0; e < 8; ++e) Bs[buf][bnn[u] + e][bkk[u]] = breg[u][e];
  };

  issueA(0, 0);
  loadB(0);
  scatterB(0);
  async_join();
  __syncthreads();

  for (int kt = 0; kt < nk; ++kt) {
    const int cur = kt & 1;
    const bool more = (kt + 1) < nk;
    if (more) { issueA(cur ^ 1, (kt + 1) * 32); loadB((kt + 1) * 32); }

    bf16x16 af[2], bfr[4];
#pragma unroll
    for (int i = 0; i < 2; ++i)
      af[i] = load_frag_a(&As[cur][wm + i * 16][0], 40, lane);
#pragma unroll
    for (int j = 0; j < 4; ++j)
      bfr[j] = load_frag_b(&Bs[cur][wn + j * 16][0], 40, lane);
#pragma unroll
    for (int i = 0; i < 2; ++i)
#pragma unroll
      for (int j = 0; j < 4; ++j) acc[i][j] = wmma_bf16(af[i], bfr[j], acc[i][j]);

    if (more) scatterB(cur ^ 1);
    async_join();
    __syncthreads();
  }

  const int hi = lane >> 4, ncol = lane & 15;
#pragma unroll
  for (int i = 0; i < 2; ++i)
#pragma unroll
    for (int j = 0; j < 4; ++j)
#pragma unroll
      for (int r = 0; r < 8; ++r) {
        int row = m0 + wm + i * 16 + r + 8 * hi;
        int col = n0 + wn + j * 16 + ncol;
        C[(size_t)row * N + col] = acc[i][j][r];
      }
}

// ---------------- RoPE + head transpose -> bf16 [B,H,S,HD] -------------------

__global__ void prep_qkv_kernel(const float* __restrict__ qlin,
                                const float* __restrict__ klin,
                                const float* __restrict__ vlin,
                                const float* __restrict__ cosT,
                                const float* __restrict__ sinT,
                                __bf16* __restrict__ qb, __bf16* __restrict__ kb,
                                __bf16* __restrict__ vb) {
  int i = blockIdx.x * blockDim.x + threadIdx.x;  // B*S*H*32
  if (i >= 2 * S_LEN * NHEAD * 32) return;
  int d2 = i & 31;
  int t = i >> 5;
  int h = t & (NHEAD - 1);
  t >>= 4;
  int s = t & (S_LEN - 1);
  int b = t >> 12;
  float cc = cosT[s * 32 + d2], ss = sinT[s * 32 + d2];
  size_t li = ((size_t)(b * S_LEN + s)) * DIM_ + h * HDIM;
  size_t o = ((size_t)((b * NHEAD + h) * S_LEN + s)) * HDIM;
  float q1 = qlin[li + d2], q2 = qlin[li + d2 + 32];
  qb[o + d2] = (__bf16)(q1 * cc - q2 * ss);
  qb[o + d2 + 32] = (__bf16)(q1 * ss + q2 * cc);
  float k1 = klin[li + d2], k2 = klin[li + d2 + 32];
  kb[o + d2] = (__bf16)(k1 * cc - k2 * ss);
  kb[o + d2 + 32] = (__bf16)(k1 * ss + k2 * cc);
  vb[o + d2] = (__bf16)vlin[li + d2];
  vb[o + d2 + 32] = (__bf16)vlin[li + d2 + 32];
}

// ---------------- block_repr: f32 mean of rotated K per block ----------------

__global__ void block_repr_kernel(const float* __restrict__ klin,
                                  const float* __restrict__ cosT,
                                  const float* __restrict__ sinT,
                                  float* __restrict__ repr) {
  int i = blockIdx.x * blockDim.x + threadIdx.x;  // B*H*NBLK*HDIM = 16384
  if (i >= 2 * NHEAD * NBLK * HDIM) return;
  int d = i & 63;
  int t = i >> 6;
  int n = t & 7;
  t >>= 3;
  int h = t & (NHEAD - 1);
  int b = t >> 4;
  int d2 = d & 31;
  bool hih = d >= 32;
  float sum = 0.f;
  for (int s = n * BLKSZ; s < (n + 1) * BLKSZ; ++s) {
    float cc = cosT[s * 32 + d2], ss = sinT[s * 32 + d2];
    size_t li = ((size_t)(b * S_LEN + s)) * DIM_ + h * HDIM;
    float k1 = klin[li + d2], k2 = klin[li + d2 + 32];
    sum += hih ? (k1 * ss + k2 * cc) : (k1 * cc - k2 * ss);
  }
  repr[i] = sum * (1.0f / (float)BLKSZ);
}

// ---------------- gating + top-3 at chunk mids (f32, stable ties) ------------

__global__ void topk_sel_kernel(const float* __restrict__ qlin,
                                const float* __restrict__ cosT,
                                const float* __restrict__ sinT,
                                const float* __restrict__ repr,
                                int* __restrict__ sel) {
  int i = blockIdx.x * blockDim.x + threadIdx.x;  // B*H*nc = 256
  if (i >= 2 * NHEAD * NBLK) return;
  int c = i & 7;
  int t = i >> 3;
  int h = t & (NHEAD - 1);
  int b = t >> 4;
  int mid = c * BLKSZ + BLKSZ / 2;
  float qr[HDIM];
  size_t li = ((size_t)(b * S_LEN + mid)) * DIM_ + h * HDIM;
  for (int d2 = 0; d2 < 32; ++d2) {
    float cc = cosT[mid * 32 + d2], ss = sinT[mid * 32 + d2];
    float q1 = qlin[li + d2], q2 = qlin[li + d2 + 32];
    qr[d2] = q1 * cc - q2 * ss;
    qr[d2 + 32] = q1 * ss + q2 * cc;
  }
  float g[NBLK];
  for (int n = 0; n < NBLK; ++n) {
    if (n > c) { g[n] = -10000.0f; continue; }
    const float* rp = &repr[((size_t)((b * NHEAD + h) * NBLK + n)) * HDIM];
    float dot = 0.f;
    for (int d = 0; d < HDIM; ++d) dot += qr[d] * rp[d];
    g[n] = dot * 0.125f;  // 1/sqrt(64)
  }
  for (int tk = 0; tk < TOPK_; ++tk) {
    float best = -3.0e38f;
    int bi = 0;
    for (int n = 0; n < NBLK; ++n)
      if (g[n] > best) { best = g[n]; bi = n; }  // strict >: stable like lax.top_k
    sel[i * TOPK_ + tk] = bi;
    g[bi] = -3.0e38f;
  }
}

// ---------------- flash attention over 3 selected blocks ---------------------
// grid: (B*H*nc, 8) ; 128 threads = 4 waves ; wave w owns query rows w*16..+15
// K/V double-buffered; V staged transposed so PV B-frags are contiguous b128.

__global__ __launch_bounds__(128) void moba_attn_kernel(
    const __bf16* __restrict__ qb, const __bf16* __restrict__ kb,
    const __bf16* __restrict__ vb, const int* __restrict__ sel,
    __bf16* __restrict__ attn) {
  __shared__ __align__(16) __bf16 Qs[64][72];
  __shared__ __align__(16) __bf16 Ks[2][64][72];   // [key][d]
  __shared__ __align__(16) __bf16 Vts[2][64][72];  // transposed [d][key]
  __shared__ __align__(16) __bf16 Ps[64][72];

  const int bhc = blockIdx.x;
  const int c = bhc & 7;
  const int bh = bhc >> 3;  // b*16 + h
  const int mtile = blockIdx.y;
  const int tid = threadIdx.x, lane = tid & 31, wave = tid >> 5;
  const int hi = lane >> 4, ncol = lane & 15;
  const size_t bh_off = (size_t)bh * S_LEN * HDIM;
  const int qbase = c * BLKSZ + mtile * 64;
  const float scale = 0.125f;
  const int NIT = TOPK_ * 8;  // 24 chunks of 64 keys

  int lrow[4], loff[4];
#pragma unroll
  for (int u = 0; u < 4; ++u) {
    int ci = tid + u * 128;
    lrow[u] = ci >> 3;
    loff[u] = (ci & 7) * 8;
  }

  bf16x8 vreg[4];
  auto issueK = [&](int buf, int krow0) {
#pragma unroll
    for (int u = 0; u < 4; ++u)
      g2l_b128(&kb[bh_off + (size_t)(krow0 + lrow[u]) * HDIM + loff[u]],
               &Ks[buf][lrow[u]][loff[u]]);
  };
  auto loadV = [&](int krow0) {
#pragma unroll
    for (int u = 0; u < 4; ++u)
      vreg[u] = *(const bf16x8*)&vb[bh_off + (size_t)(krow0 + lrow[u]) * HDIM + loff[u]];
  };
  auto scatterV = [&](int buf) {
#pragma unroll
    for (int u = 0; u < 4; ++u)
#pragma unroll
      for (int e = 0; e < 8; ++e) Vts[buf][loff[u] + e][lrow[u]] = vreg[u][e];
  };

  // Q tile (async) + first K/V chunk
#pragma unroll
  for (int u = 0; u < 4; ++u)
    g2l_b128(&qb[bh_off + (size_t)(qbase + lrow[u]) * HDIM + loff[u]],
             &Qs[lrow[u]][loff[u]]);
  const int blk0 = sel[bhc * TOPK_];
  issueK(0, blk0 * BLKSZ);
  loadV(blk0 * BLKSZ);
  scatterV(0);
  async_join();
  __syncthreads();

  // loop-invariant Q fragments
  const bf16x16 aq0 = load_frag_a(&Qs[wave * 16][0], 72, lane);
  const bf16x16 aq1 = load_frag_a(&Qs[wave * 16][32], 72, lane);

  float mr[8], lr[8];
#pragma unroll
  for (int r = 0; r < 8; ++r) { mr[r] = -3.0e38f; lr[r] = 0.f; }
  f32x8 o[4] = {};

  for (int it = 0; it < NIT; ++it) {
    const int cur = it & 1;
    const bool more = (it + 1) < NIT;
    if (more) {
      int nxt = it + 1;
      int krow0 = sel[bhc * TOPK_ + (nxt >> 3)] * BLKSZ + (nxt & 7) * 64;
      issueK(cur ^ 1, krow0);
      loadV(krow0);
    }

    // S = Q Kt (scaled): 4 key subtiles x 2 k-steps over HD=64
    f32x8 sacc[4];
#pragma unroll
    for (int nt = 0; nt < 4; ++nt) {
      f32x8 z = {};
      bf16x16 b0 = load_frag_b(&Ks[cur][nt * 16][0], 72, lane);
      bf16x16 b1 = load_frag_b(&Ks[cur][nt * 16][32], 72, lane);
      z = wmma_bf16(aq0, b0, z);
      z = wmma_bf16(aq1, b1, z);
      sacc[nt] = z;
    }

    // online softmax (per-lane row state: row = r + 8*hi)
#pragma unroll
    for (int r = 0; r < 8; ++r) {
      float v = -3.0e38f;
#pragma unroll
      for (int nt = 0; nt < 4; ++nt) {
        sacc[nt][r] *= scale;
        v = fmaxf(v, sacc[nt][r]);
      }
      v = fmaxf(v, __shfl_xor(v, 1));
      v = fmaxf(v, __shfl_xor(v, 2));
      v = fmaxf(v, __shfl_xor(v, 4));
      v = fmaxf(v, __shfl_xor(v, 8));
      float mn = fmaxf(mr[r], v);
      float al = __expf(mr[r] - mn);
      float ps = 0.f;
#pragma unroll
      for (int nt = 0; nt < 4; ++nt) {
        float p = __expf(sacc[nt][r] - mn);
        sacc[nt][r] = p;
        ps += p;
      }
      ps += __shfl_xor(ps, 1);
      ps += __shfl_xor(ps, 2);
      ps += __shfl_xor(ps, 4);
      ps += __shfl_xor(ps, 8);
      lr[r] = lr[r] * al + ps;
      mr[r] = mn;
#pragma unroll
      for (int dt = 0; dt < 4; ++dt) o[dt][r] *= al;
    }

    // spill P (own rows only -> same-wave LDS ordering suffices)
#pragma unroll
    for (int nt = 0; nt < 4; ++nt)
#pragma unroll
      for (int r = 0; r < 8; ++r)
        Ps[wave * 16 + r + 8 * hi][nt * 16 + ncol] = (__bf16)sacc[nt][r];

    // O += P * V  (contiguous B-frags from transposed V)
    bf16x16 ap0 = load_frag_a(&Ps[wave * 16][0], 72, lane);
    bf16x16 ap1 = load_frag_a(&Ps[wave * 16][32], 72, lane);
#pragma unroll
    for (int dt = 0; dt < 4; ++dt) {
      bf16x16 b0 = load_frag_b(&Vts[cur][dt * 16][0], 72, lane);
      bf16x16 b1 = load_frag_b(&Vts[cur][dt * 16][32], 72, lane);
      o[dt] = wmma_bf16(ap0, b0, o[dt]);
      o[dt] = wmma_bf16(ap1, b1, o[dt]);
    }

    if (more) scatterV(cur ^ 1);
    async_join();
    __syncthreads();
  }

  // epilogue: normalize and write attn [B*S, DIM] bf16 (head-interleaved)
  const int b = bh >> 4, h = bh & (NHEAD - 1);
#pragma unroll
  for (int dt = 0; dt < 4; ++dt)
#pragma unroll
    for (int r = 0; r < 8; ++r) {
      int row = qbase + wave * 16 + r + 8 * hi;
      float val = o[dt][r] / lr[r];
      attn[((size_t)(b * S_LEN + row)) * DIM_ + h * HDIM + dt * 16 + ncol] =
          (__bf16)val;
    }
}

// ---------------- host-side orchestration ------------------------------------

extern "C" void kernel_launch(void* const* d_in, const int* in_sizes, int n_in,
                              void* d_out, int out_size, void* d_ws, size_t ws_size,
                              hipStream_t stream) {
  (void)in_sizes; (void)n_in; (void)out_size; (void)ws_size;
  const float* x = (const float*)d_in[0];
  const float* wq = (const float*)d_in[1];
  const float* wk = (const float*)d_in[2];
  const float* wv = (const float*)d_in[3];
  const float* wo = (const float*)d_in[4];
  const float* cosT = (const float*)d_in[5];
  const float* sinT = (const float*)d_in[6];
  float* out = (float*)d_out;

  char* ws = (char*)d_ws;
  size_t off = 0;
  auto alloc = [&](size_t bytes) -> void* {
    void* p = ws + off;
    off = (off + bytes + 255) & ~(size_t)255;
    return p;
  };

  const size_t NX = (size_t)MROWS * DIM_;  // 8,388,608
  const size_t NW = (size_t)DIM_ * DIM_;   // 1,048,576
  __bf16* xh  = (__bf16*)alloc(NX * 2);
  __bf16* wqh = (__bf16*)alloc(NW * 2);
  __bf16* wkh = (__bf16*)alloc(NW * 2);
  __bf16* wvh = (__bf16*)alloc(NW * 2);
  __bf16* woh = (__bf16*)alloc(NW * 2);
  float* qlin = (float*)alloc(NX * 4);
  float* klin = (float*)alloc(NX * 4);
  float* vlin = (float*)alloc(NX * 4);
  __bf16* qb = (__bf16*)alloc(NX * 2);
  __bf16* kb = (__bf16*)alloc(NX * 2);
  __bf16* vb = (__bf16*)alloc(NX * 2);
  float* repr = (float*)alloc((size_t)2 * NHEAD * NBLK * HDIM * 4);
  int* sel = (int*)alloc((size_t)2 * NHEAD * NBLK * TOPK_ * 4);
  __bf16* attnb = (__bf16*)alloc(NX * 2);

  f32_to_bf16_kernel<<<(int)(NX / 8 / 256), 256, 0, stream>>>(x, xh, (int)NX);
  f32_to_bf16_kernel<<<(int)(NW / 8 / 256), 256, 0, stream>>>(wq, wqh, (int)NW);
  f32_to_bf16_kernel<<<(int)(NW / 8 / 256), 256, 0, stream>>>(wk, wkh, (int)NW);
  f32_to_bf16_kernel<<<(int)(NW / 8 / 256), 256, 0, stream>>>(wv, wvh, (int)NW);
  f32_to_bf16_kernel<<<(int)(NW / 8 / 256), 256, 0, stream>>>(wo, woh, (int)NW);

  dim3 gg(MROWS / 128, DIM_ / 128);  // 64 x 8
  gemm_bf16_kernel<<<gg, 256, 0, stream>>>(xh, wqh, qlin, MROWS, DIM_, DIM_);
  gemm_bf16_kernel<<<gg, 256, 0, stream>>>(xh, wkh, klin, MROWS, DIM_, DIM_);
  gemm_bf16_kernel<<<gg, 256, 0, stream>>>(xh, wvh, vlin, MROWS, DIM_, DIM_);

  prep_qkv_kernel<<<(2 * S_LEN * NHEAD * 32) / 256, 256, 0, stream>>>(
      qlin, klin, vlin, cosT, sinT, qb, kb, vb);
  block_repr_kernel<<<(2 * NHEAD * NBLK * HDIM + 255) / 256, 256, 0, stream>>>(
      klin, cosT, sinT, repr);
  topk_sel_kernel<<<1, 256, 0, stream>>>(qlin, cosT, sinT, repr, sel);

  moba_attn_kernel<<<dim3(2 * NHEAD * NBLK, 8), 128, 0, stream>>>(qb, kb, vb, sel,
                                                                  attnb);

  gemm_bf16_kernel<<<gg, 256, 0, stream>>>(attnb, woh, out, MROWS, DIM_, DIM_);
}
